// DecoderBlock_74002286510082
// MI455X (gfx1250) — matrix operand order repaired and checked by reading, not compiled, not run on hardware
//
#include <hip/hip_runtime.h>
#include <hip/hip_bf16.h>

// ---------------------------------------------------------------------------
// Types
// ---------------------------------------------------------------------------
typedef __bf16 bf16_t;
typedef __attribute__((ext_vector_type(16))) __bf16 v16bf;
typedef __attribute__((ext_vector_type(8)))  __bf16 v8bf;
typedef __attribute__((ext_vector_type(8)))  float  v8f;

// Problem constants (match reference)
#define BB   2
#define SS   2048
#define HH   512
#define NHD  8
#define DD   4096          // NHD * HH (per source quirk: head_size == H)
#define D3   12288         // 3 * DD
#define ROWS 4096          // BB * SS

// Block-tile geometry for the WMMA GEMM
#define BM   64            // block tile M
#define BN   128           // block tile N
#define BK   32            // k-step
#define KPAD 40            // padded k-stride in LDS (80B; 16B aligned, bank-spread)

__device__ __forceinline__ float bf2f(bf16_t v) { return (float)v; }
__device__ __forceinline__ bf16_t f2bf(float f) { return (bf16_t)f; }

// LDS byte-offset of a __shared__ object: low 32 bits of its generic address.
__device__ __forceinline__ unsigned lds_off_of(const void* p) {
  return (unsigned)(unsigned long long)p;
}

// gfx1250 async copy global -> LDS (16B per lane), tracked by ASYNCcnt.
__device__ __forceinline__ void async_load_b128(unsigned lds_off, const void* gptr) {
  asm volatile("global_load_async_to_lds_b128 %0, %1, off"
               :: "v"(lds_off), "v"(gptr) : "memory");
}

// ---------------------------------------------------------------------------
// f32 -> bf16 elementwise conversion (grid-stride)
// ---------------------------------------------------------------------------
__global__ void cvt_f32_bf16(const float* __restrict__ in,
                             bf16_t* __restrict__ out, long n) {
  long i = (long)blockIdx.x * blockDim.x + threadIdx.x;
  long stride = (long)gridDim.x * blockDim.x;
  for (; i < n; i += stride) out[i] = f2bf(in[i]);
}

// ---------------------------------------------------------------------------
// LayerNorm over last dim (H=512), one block per row, bf16 output
// ---------------------------------------------------------------------------
__global__ void ln_to_bf16(const float* __restrict__ x,
                           const float* __restrict__ g,
                           const float* __restrict__ b,
                           bf16_t* __restrict__ out) {
  __shared__ float s_sum[256];
  __shared__ float s_sq[256];
  const long row = blockIdx.x;
  const float* xr = x + row * (long)HH;
  float sum = 0.f, sq = 0.f;
  for (int i = threadIdx.x; i < HH; i += blockDim.x) {
    float v = xr[i];
    sum += v; sq += v * v;
  }
  s_sum[threadIdx.x] = sum; s_sq[threadIdx.x] = sq;
  __syncthreads();
  for (int s = blockDim.x >> 1; s > 0; s >>= 1) {
    if ((int)threadIdx.x < s) {
      s_sum[threadIdx.x] += s_sum[threadIdx.x + s];
      s_sq[threadIdx.x]  += s_sq[threadIdx.x + s];
    }
    __syncthreads();
  }
  const float mean = s_sum[0] * (1.0f / HH);
  const float var  = s_sq[0] * (1.0f / HH) - mean * mean;
  const float inv  = rsqrtf(var + 1e-5f);
  for (int i = threadIdx.x; i < HH; i += blockDim.x) {
    out[row * (long)HH + i] = f2bf((xr[i] - mean) * inv * g[i] + b[i]);
  }
}

// ---------------------------------------------------------------------------
// LDS-tiled transpose of V: qkv[b, t, 2D + h*hs + c] -> vt[(b*NH+h), c, t]
// ---------------------------------------------------------------------------
__global__ void transpose_v(const bf16_t* __restrict__ qkv,
                            bf16_t* __restrict__ vt) {
  __shared__ bf16_t tile[32][33];
  const int z  = blockIdx.z;            // b*NH + h
  const int bb = z / NHD;
  const int hh = z % NHD;
  const long t0 = (long)blockIdx.x * 32;
  const long c0 = (long)blockIdx.y * 32;
  const bf16_t* src = qkv + (long)bb * SS * D3 + 2 * DD + (long)hh * HH;
#pragma unroll
  for (int i = 0; i < 4; ++i) {
    const int tl = threadIdx.y * 4 + i;           // local t
    tile[tl][threadIdx.x] = src[(t0 + tl) * (long)D3 + c0 + threadIdx.x];
  }
  __syncthreads();
  bf16_t* dst = vt + ((long)z * HH + c0) * SS + t0;
#pragma unroll
  for (int i = 0; i < 4; ++i) {
    const int cl = threadIdx.y * 4 + i;           // local c
    dst[(long)cl * SS + threadIdx.x] = tile[threadIdx.x][cl];
  }
}

// ---------------------------------------------------------------------------
// Softmax over the HEADS axis (axis=1), in-place on bf16 scores [B,NH,S,S]
// ---------------------------------------------------------------------------
__global__ void softmax_heads(bf16_t* __restrict__ sc) {
  const long total = (long)BB * SS * SS;
  long idx = (long)blockIdx.x * blockDim.x + threadIdx.x;
  if (idx >= total) return;
  const long t = idx % SS;
  const long s = (idx / SS) % SS;
  const long b = idx / ((long)SS * SS);
  const long hstride = (long)SS * SS;
  bf16_t* base = sc + b * (long)NHD * hstride + s * (long)SS + t;
  float v[NHD];
  float mx = -3.0e38f;
#pragma unroll
  for (int h = 0; h < NHD; ++h) {
    v[h] = bf2f(base[h * hstride]);
    mx = fmaxf(mx, v[h]);
  }
  float denom = 0.f;
#pragma unroll
  for (int h = 0; h < NHD; ++h) {
    v[h] = __expf(v[h] - mx);
    denom += v[h];
  }
  const float inv = 1.0f / denom;
#pragma unroll
  for (int h = 0; h < NHD; ++h) base[h * hstride] = f2bf(v[h] * inv);
}

// ---------------------------------------------------------------------------
// WMMA helpers
// ---------------------------------------------------------------------------
__device__ __forceinline__ v8f wmma_bf16(v16bf a, v16bf b, v8f c) {
  return __builtin_amdgcn_wmma_f32_16x16x32_bf16(
      /*neg_a=*/false, a, /*neg_b=*/false, b,
      /*c_mod=*/(short)0, c, /*reuse_a=*/false, /*reuse_b=*/false);
}

// A 16x32 fragment from an LDS row (row = 32 valid bf16, KPAD stride handled
// by caller): lanes 0-15 -> K {0..7,16..23}; lanes 16-31 -> K {8..15,24..31}
__device__ __forceinline__ v16bf lds_a_frag(const bf16_t* row, int half) {
  const v8bf lo = *(const v8bf*)(row + half * 8);
  const v8bf hi = *(const v8bf*)(row + half * 8 + 16);
  v16bf a;
#pragma unroll
  for (int j = 0; j < 8; ++j) { a[j] = lo[j]; a[j + 8] = hi[j]; }
  return a;
}

// B 32x16 fragment: lane's column, K = half*8 + {0..7, 16..23} (two 16B LDS loads)
__device__ __forceinline__ v16bf lds_b_frag(const bf16_t* col, int half) {
  const v8bf lo = *(const v8bf*)(col + half * 8);
  const v8bf hi = *(const v8bf*)(col + half * 8 + 16);
  v16bf b;
#pragma unroll
  for (int j = 0; j < 8; ++j) { b[j] = lo[j]; b[j + 8] = hi[j]; }
  return b;
}

// Epilogues:
//   0: store bf16
//   1: triu mask (keep n>=m) * scale, store bf16          (attention scores)
//   2: +bias, exact GELU, store bf16                       (MLP hidden)
//   3: +bias, store bf16                                   (QKV)
//   4: +bias, +residual, store f32                         (Wo / final out)
template <int EPI>
__device__ __forceinline__ void store_tile(
    v8f acc, void* __restrict__ Cp, long coff, long cm,
    long mbase, long nbase, int half, int lid,
    const float* __restrict__ bias, const float* __restrict__ resid,
    long rm, float scale) {
#pragma unroll
  for (int v = 0; v < 8; ++v) {
    const long m = mbase + half * 8 + v;
    const long n = nbase + lid;
    float val = acc[v];
    if (EPI == 1) val = (n >= m) ? val * scale : 0.0f;
    if (EPI >= 2) val += bias[n];
    if (EPI == 2) val = 0.5f * val * (1.0f + erff(val * 0.70710678118654752f));
    if (EPI == 4) {
      ((float*)Cp)[coff + m * cm + n] = val + resid[m * rm + n];
    } else {
      ((bf16_t*)Cp)[coff + m * cm + n] = f2bf(val);
    }
  }
}

// ---------------------------------------------------------------------------
// Batched WMMA GEMM with async-to-LDS double-buffered staging.
//   C[m,n] = sum_k A[m,k] * B[n,k]  (both operands K-contiguous)
//   A[m,k] at A + m*am + k ;  B[n,k] at Bm + n*bn + k
//   grid: x = N/BN, y = M/BM, z = batches (z -> (b,h) via nh)
//   block: 256 threads = 8 waves (2 M x 4 N), wave tile 32x32 (2x2 wmma tiles)
// ---------------------------------------------------------------------------
template <int EPI>
__global__ __launch_bounds__(256)
void wmma_gemm(
    const bf16_t* __restrict__ A,  long am, long a_bs, long a_hs,
    const bf16_t* __restrict__ Bm, long bn, long b_bs, long b_hs,
    void* __restrict__ Cp,         long cm, long c_bs, long c_hs,
    const float* __restrict__ bias,
    const float* __restrict__ resid, long rm,
    int K, int nh, float scale) {
  __shared__ bf16_t At[2][BM][KPAD];   // 2 * 64*40*2  = 10.2 KB
  __shared__ bf16_t Bt[2][BN][KPAD];   // 2 * 128*40*2 = 20.5 KB

  const int z  = blockIdx.z;
  const int bb = z / nh;
  const int hh = z % nh;
  const bf16_t* Ab = A  + (long)bb * a_bs + (long)hh * a_hs;
  const bf16_t* Bb = Bm + (long)bb * b_bs + (long)hh * b_hs;
  const long coff = (long)bb * c_bs + (long)hh * c_hs;

  const int tid = threadIdx.x;
  const long m0blk = (long)blockIdx.y * BM;
  const long n0blk = (long)blockIdx.x * BN;

  // --- staging coordinates: A 64x32 (1 x b128/thread), B 128x32 (2 x b128) ---
  const int arow  = tid >> 2;            // 0..63
  const int akoff = (tid & 3) * 8;       // 0,8,16,24
  const int bcol  = tid >> 1;            // 0..127
  const int bkoff = (tid & 1) * 16;      // 0,16
  const bf16_t* Aload = Ab + (m0blk + arow) * am + akoff;
  const bf16_t* Bload = Bb + (n0blk + bcol) * bn + bkoff;
  const unsigned At_o = lds_off_of(&At[0][arow][akoff]);
  const unsigned Bt_o = lds_off_of(&Bt[0][bcol][bkoff]);
  const unsigned At_stride = sizeof(At[0]);
  const unsigned Bt_stride = sizeof(Bt[0]);

  // --- wave / lane coordinates ---
  const int w    = tid >> 5;
  const int wm   = w >> 2;               // 0..1
  const int wn   = w & 3;                // 0..3
  const int lane = tid & 31;
  const int half = lane >> 4;
  const int lid  = lane & 15;
  const int ar0  = wm * 32 + lid;        // A rows ar0, ar0+16
  const int bc0  = wn * 32 + lid;        // B cols bc0, bc0+16

  v8f acc00 = {}, acc01 = {}, acc10 = {}, acc11 = {};
  const int nk = K >> 5;

  auto issue = [&](int kt, int buf) {
    const long k0 = (long)kt * BK;
    async_load_b128(At_o + buf * At_stride, Aload + k0);
    async_load_b128(Bt_o + buf * Bt_stride, Bload + k0);
    async_load_b128(Bt_o + buf * Bt_stride + 16, Bload + k0 + 8);
  };

  issue(0, 0);
  for (int ki = 0; ki < nk; ++ki) {
    const int cur = ki & 1;
    if (ki + 1 < nk) {
      issue(ki + 1, cur ^ 1);
      asm volatile("s_wait_asynccnt 0x3" ::: "memory");  // tile ki complete
    } else {
      asm volatile("s_wait_asynccnt 0x0" ::: "memory");
    }
    __syncthreads();

    const v16bf a0 = lds_a_frag(&At[cur][ar0][0],      half);
    const v16bf a1 = lds_a_frag(&At[cur][ar0 + 16][0], half);
    const v16bf b0 = lds_b_frag(&Bt[cur][bc0][0],      half);
    const v16bf b1 = lds_b_frag(&Bt[cur][bc0 + 16][0], half);
    acc00 = wmma_bf16(a0, b0, acc00);
    acc01 = wmma_bf16(a0, b1, acc01);
    acc10 = wmma_bf16(a1, b0, acc10);
    acc11 = wmma_bf16(a1, b1, acc11);

    __syncthreads();   // all waves done reading buf[cur] before it is refilled
  }

  const long m0 = m0blk + wm * 32;
  const long n0 = n0blk + wn * 32;
  store_tile<EPI>(acc00, Cp, coff, cm, m0,      n0,      half, lid, bias, resid, rm, scale);
  store_tile<EPI>(acc01, Cp, coff, cm, m0,      n0 + 16, half, lid, bias, resid, rm, scale);
  store_tile<EPI>(acc10, Cp, coff, cm, m0 + 16, n0,      half, lid, bias, resid, rm, scale);
  store_tile<EPI>(acc11, Cp, coff, cm, m0 + 16, n0 + 16, half, lid, bias, resid, rm, scale);
}

// ---------------------------------------------------------------------------
// Launch
// ---------------------------------------------------------------------------
extern "C" void kernel_launch(void* const* d_in, const int* in_sizes, int n_in,
                              void* d_out, int out_size, void* d_ws, size_t ws_size,
                              hipStream_t stream) {
  const float* x      = (const float*)d_in[0];
  const float* ln1_g  = (const float*)d_in[2];
  const float* ln1_b  = (const float*)d_in[3];
  const float* wqkv_w = (const float*)d_in[4];
  const float* wqkv_b = (const float*)d_in[5];
  const float* wo_w   = (const float*)d_in[6];
  const float* wo_b   = (const float*)d_in[7];
  const float* ln2_g  = (const float*)d_in[8];
  const float* ln2_b  = (const float*)d_in[9];
  const float* w1     = (const float*)d_in[10];
  const float* b1     = (const float*)d_in[11];
  const float* w2     = (const float*)d_in[12];
  const float* b2     = (const float*)d_in[13];
  float* out = (float*)d_out;

  // ---- workspace carve-out (256B aligned) ----
  size_t off = 0;
  auto carve = [&](size_t bytes) -> void* {
    void* p = (char*)d_ws + off;
    off += (bytes + 255) & ~(size_t)255;
    return p;
  };
  bf16_t* wqkv_bf = (bf16_t*)carve((size_t)D3 * HH * 2);            // 12.6 MB
  bf16_t* wo_bf   = (bf16_t*)carve((size_t)HH * DD * 2);            //  4.2 MB
  bf16_t* w1_bf   = (bf16_t*)carve((size_t)4 * HH * HH * 2);        //  2.1 MB
  bf16_t* w2_bf   = (bf16_t*)carve((size_t)HH * 4 * HH * 2);        //  2.1 MB
  bf16_t* h_bf    = (bf16_t*)carve((size_t)ROWS * HH * 2);          //  4.2 MB (LN1/LN2)
  bf16_t* qkv_bf  = (bf16_t*)carve((size_t)ROWS * D3 * 2);          // 100.7 MB
  bf16_t* vt_bf   = (bf16_t*)carve((size_t)BB * NHD * HH * SS * 2); // 33.6 MB (V^T)
  bf16_t* o_bf    = (bf16_t*)carve((size_t)ROWS * DD * 2);          // 33.6 MB
  float*  x1      = (float*)carve((size_t)ROWS * HH * 4);           //  8.4 MB
  bf16_t* sc_bf   = (bf16_t*)carve((size_t)BB * NHD * SS * SS * 2); // 134.2 MB
  bf16_t* m1_bf   = sc_bf;  // MLP hidden aliases scores (dead after A·V)

  const dim3 blk256(256);

  // ---- weights -> bf16 ----
  cvt_f32_bf16<<<4096, blk256, 0, stream>>>(wqkv_w, wqkv_bf, (long)D3 * HH);
  cvt_f32_bf16<<<2048, blk256, 0, stream>>>(wo_w,   wo_bf,   (long)HH * DD);
  cvt_f32_bf16<<<1024, blk256, 0, stream>>>(w1,     w1_bf,   (long)4 * HH * HH);
  cvt_f32_bf16<<<1024, blk256, 0, stream>>>(w2,     w2_bf,   (long)HH * 4 * HH);

  // ---- LN1 ----
  ln_to_bf16<<<ROWS, blk256, 0, stream>>>(x, ln1_g, ln1_b, h_bf);

  // ---- QKV GEMM: [4096,512] x [12288,512]^T -> bf16 [4096,12288], +bias ----
  wmma_gemm<3><<<dim3(D3 / BN, ROWS / BM, 1), blk256, 0, stream>>>(
      h_bf, HH, 0, 0,
      wqkv_bf, HH, 0, 0,
      qkv_bf, D3, 0, 0,
      wqkv_b, nullptr, 0,
      HH, 1, 1.0f);

  // ---- transpose V into [B*NH, hs, S] so A·V hits the contiguous-B path ----
  transpose_v<<<dim3(SS / 32, HH / 32, BB * NHD), dim3(32, 8), 0, stream>>>(
      qkv_bf, vt_bf);

  // ---- scores = Q K^T per (b,h): triu-mask * 1/sqrt(4096) ----
  wmma_gemm<1><<<dim3(SS / BN, SS / BM, BB * NHD), blk256, 0, stream>>>(
      qkv_bf,      D3, (long)SS * D3, HH,                 // A = Q
      qkv_bf + DD, D3, (long)SS * D3, HH,                 // B = K
      sc_bf,       SS, (long)NHD * SS * SS, (long)SS * SS,
      nullptr, nullptr, 0,
      HH, NHD, 0.015625f /* 1/sqrt(4096) */);

  // ---- softmax over heads, in place ----
  {
    long total = (long)BB * SS * SS;
    softmax_heads<<<(unsigned)((total + 255) / 256), blk256, 0, stream>>>(sc_bf);
  }

  // ---- o = A V per (b,h): [S,S] x [S,512] -> bf16 [B,S,4096] ----
  wmma_gemm<0><<<dim3(HH / BN, SS / BM, BB * NHD), blk256, 0, stream>>>(
      sc_bf, SS, (long)NHD * SS * SS, (long)SS * SS,      // A = probs
      vt_bf, SS, (long)NHD * HH * SS, (long)HH * SS,      // B = V^T (K-contiguous)
      o_bf,  DD, (long)SS * DD, HH,
      nullptr, nullptr, 0,
      SS, NHD, 1.0f);

  // ---- attn out: [4096,4096] x [512,4096]^T + bias + residual(x) -> f32 x1 ----
  wmma_gemm<4><<<dim3(HH / BN, ROWS / BM, 1), blk256, 0, stream>>>(
      o_bf, DD, 0, 0,
      wo_bf, DD, 0, 0,
      x1, HH, 0, 0,
      wo_b, x, HH,
      DD, 1, 1.0f);

  // ---- LN2 ----
  ln_to_bf16<<<ROWS, blk256, 0, stream>>>(x1, ln2_g, ln2_b, h_bf);

  // ---- MLP1: [4096,512] x [2048,512]^T + bias, exact GELU -> bf16 ----
  wmma_gemm<2><<<dim3((4 * HH) / BN, ROWS / BM, 1), blk256, 0, stream>>>(
      h_bf, HH, 0, 0,
      w1_bf, HH, 0, 0,
      m1_bf, 4 * HH, 0, 0,
      b1, nullptr, 0,
      HH, 1, 1.0f);

  // ---- MLP2: [4096,2048] x [512,2048]^T + bias + residual(x1) -> f32 out ----
  wmma_gemm<4><<<dim3(HH / BN, ROWS / BM, 1), blk256, 0, stream>>>(
      m1_bf, 4 * HH, 0, 0,
      w2_bf, 4 * HH, 0, 0,
      out, HH, 0, 0,
      b2, x1, HH,
      4 * HH, 1, 1.0f);

  (void)in_sizes; (void)n_in; (void)out_size; (void)ws_size;
}
